// NeuralProgramSynthesis_77799037599849
// MI455X (gfx1250) — matrix operand order repaired and checked by reading, not compiled, not run on hardware
//
#include <hip/hip_runtime.h>
#include <hip/hip_bf16.h>

// ---------------------------------------------------------------------------
// MI455X (gfx1250) implementation.
// ~430 GFLOP of expert-conv work vs ~340MB HBM traffic @23.3TB/s -> compute
// bound -> all heavy math via v_wmma_f32_16x16x32_bf16. Conv3x3 is implicit
// GEMM with D = W^T x patch so conv1's f32 C/D register layout IS conv2's
// bf16 A-fragment layout (zero cross-lane movement between the convs).
// This revision stages the conv input halo in LDS: one coalesced load per
// grid value per block, then all im2col gathers are ds_load_b32 with 32-bit
// addressing and no bounds checks (halo padding materialized in LDS).
// ---------------------------------------------------------------------------

typedef __attribute__((ext_vector_type(16))) __bf16 v16bf;
typedef __attribute__((ext_vector_type(8)))  float  v8f;

union BF16Frag {
    v16bf v;
    unsigned int u[8];
    uint4 q[2];
};

// f32 pair -> packed bf16 (round-to-nearest via +0x8000, pack via v_perm_b32)
__device__ __forceinline__ unsigned int pk_bf16(float lo, float hi) {
    unsigned int a = __builtin_bit_cast(unsigned int, lo) + 0x8000u;
    unsigned int b = __builtin_bit_cast(unsigned int, hi) + 0x8000u;
    return __builtin_amdgcn_perm(b, a, 0x07060302u);  // {b.hi16, a.hi16}
}

// branch-free tanh: 1 - 2/(e^{2x}+1); saturates correctly at +/-inf
__device__ __forceinline__ float fast_tanh(float x) {
    float e2 = __expf(2.0f * x);                       // v_exp_f32 path
    return 1.0f - 2.0f * __builtin_amdgcn_rcpf(e2 + 1.0f);
}

__device__ __forceinline__ v8f wmma_bf16(v16bf a, v16bf b, v8f c) {
    // (neg_a, A, neg_b, B, c_mod, C, reuse_a, reuse_b)
    return __builtin_amdgcn_wmma_f32_16x16x32_bf16(false, a, false, b,
                                                   (short)0, c, false, false);
}

// ---------------------------------------------------------------------------
// 1) Global average pool: features [128,30,30,512] -> g [128,512]
// ---------------------------------------------------------------------------
__global__ __launch_bounds__(256) void pool_kernel(const float* __restrict__ f,
                                                   float* __restrict__ g) {
    int id = blockIdx.x * 256 + threadIdx.x;
    if (id >= 128 * 512) return;
    int b = id >> 9, d = id & 511;
    const float* p = f + (size_t)b * 900 * 512 + d;
    float s = 0.f;
    for (int i = 0; i < 900; ++i) s += p[(size_t)i * 512];
    g[id] = s * (1.f / 900.f);
}

// ---------------------------------------------------------------------------
// 2) Generic bf16 WMMA GEMM: out[M,N] = act(A[M,K] @ B[K,N] + bias[N])
//    One wave per 16x16 output tile. act: 0 = identity, 1 = exact GELU.
// ---------------------------------------------------------------------------
__global__ __launch_bounds__(256) void gemm_bf16_kernel(
    const float* __restrict__ A, const float* __restrict__ Bm,
    const float* __restrict__ bias, float* __restrict__ out,
    int M, int N, int K, int act, int numWaves) {
    int wid = (blockIdx.x * 256 + threadIdx.x) >> 5;
    if (wid >= numWaves) return;
    int lane = threadIdx.x & 31;
    int m16 = lane & 15, g = lane >> 4;
    int mt = wid % (M / 16);
    int nt = wid / (M / 16);

    v8f acc = (v8f)0.f;
    int row = mt * 16 + m16;
    int col = nt * 16 + m16;
    for (int kb = 0; kb < K; kb += 32) {
        BF16Frag af, bfr;
#pragma unroll
        for (int v = 0; v < 8; ++v) {   // A fragment (16-bit A layout)
            int K0 = 2 * v + 16 * g;
            af.u[v] = pk_bf16(A[(size_t)row * K + kb + K0],
                              A[(size_t)row * K + kb + K0 + 1]);
        }
#pragma unroll
        for (int v = 0; v < 8; ++v) {   // B fragment (16-bit B layout)
            int Kr = kb + 2 * v + 16 * g;
            bfr.u[v] = pk_bf16(Bm[(size_t)Kr * N + col],
                               Bm[(size_t)(Kr + 1) * N + col]);
        }
        acc = wmma_bf16(af.v, bfr.v, acc);
    }
    float bv = bias[col];
#pragma unroll
    for (int r = 0; r < 8; ++r) {
        int orow = mt * 16 + r + 8 * g;
        float x = acc[r] + bv;
        if (act) x = 0.5f * x * (1.0f + erff(x * 0.70710678118654752f));
        out[(size_t)orow * N + col] = x;
    }
}

// ---------------------------------------------------------------------------
// 3) Softmax over the 16 ops of each (batch, step) group.
// ---------------------------------------------------------------------------
__global__ __launch_bounds__(256) void softmax_kernel(
    const float* __restrict__ logits, float* __restrict__ probs_ws,
    float* __restrict__ probs_out) {
    int id = blockIdx.x * 256 + threadIdx.x;
    if (id >= 128 * 8) return;
    const float* lp = logits + id * 16;
    float mx = lp[0];
    for (int i = 1; i < 16; ++i) mx = fmaxf(mx, lp[i]);
    float e[16], s = 0.f;
    for (int i = 0; i < 16; ++i) { e[i] = __expf(lp[i] - mx); s += e[i]; }
    float inv = 1.f / s;
    for (int i = 0; i < 16; ++i) {
        float p = e[i] * inv;
        probs_ws[id * 16 + i] = p;
        probs_out[id * 16 + i] = p;
    }
}

// ---------------------------------------------------------------------------
// 4) Pack expert weights into lane-exact bf16 WMMA fragments.
//    frag1: W1^T as A-fragments [e][dy][ntile] (M=outch tile, K=3taps*10ch)
//    frag2: W2   as B-fragments [e]            (K=32 hidden, N=10(pad16) out)
// ---------------------------------------------------------------------------
__global__ __launch_bounds__(256) void prep_frags(
    const float* __restrict__ W1, const float* __restrict__ W2,
    unsigned int* __restrict__ frag1, unsigned int* __restrict__ frag2) {
    int id = blockIdx.x * 256 + threadIdx.x;
    if (id < 96 * 256) {
        int v = id & 7, lane = (id >> 3) & 31, fi = id >> 8;
        int ntile = fi & 1, dy = (fi >> 1) % 3, e = fi / 6;
        int g = lane >> 4, m = lane & 15;
        int oc = ntile * 16 + m;
        int K0 = 2 * v + 16 * g;
        float x[2];
#pragma unroll
        for (int h2 = 0; h2 < 2; ++h2) {
            int K = K0 + h2;
            float val = 0.f;
            if (K < 30) {
                int c = K % 10, dx = K / 10;
                val = W1[((size_t)(e * 32 + oc) * 10 + c) * 9 + dy * 3 + dx];
            }
            x[h2] = val;
        }
        frag1[(size_t)fi * 256 + lane * 8 + v] = pk_bf16(x[0], x[1]);
    } else if (id < 96 * 256 + 16 * 256) {
        int id2 = id - 96 * 256;
        int v = id2 & 7, lane = (id2 >> 3) & 31, e = id2 >> 8;
        int n = lane & 15, g = lane >> 4;
        int Kr = ((v < 4) ? 2 * v : 8 + 2 * v) + 8 * g;
        float x0 = (n < 10) ? W2[(size_t)(e * 10 + n) * 32 + Kr] : 0.f;
        float x1 = (n < 10) ? W2[(size_t)(e * 10 + n) * 32 + Kr + 1] : 0.f;
        frag2[(size_t)e * 256 + lane * 8 + v] = pk_bf16(x0, x1);
    }
}

// ---------------------------------------------------------------------------
// 5) One program step. Block = one batch image, 2 grid rows (8 waves =
//    2 rows x 4 col-tiles). Input halo (4 rows x 10 ch x 66 cols, zero
//    padded) staged once in LDS; im2col gathers are ds_load_b32.
//    conv1: D1[outch x pos] = (W1^T)A x (patch)B  -- 6 WMMA/expert
//    conv1 D-layout == conv2 A-layout  => per-lane relu+pack only.
//    conv2: D2[pos x outch] = (h)A x (W2)B        -- 1 WMMA/expert
//    combine: out += p[b,e] * tanh(D2)            -- registers only
// ---------------------------------------------------------------------------
#define TILE_F (4 * 10 * 66)    // 2640 floats = 10.56 KB

__global__ __launch_bounds__(256) void step_kernel(
    const float* __restrict__ gin, long gin_bstride,
    const unsigned int* __restrict__ frag1, const unsigned int* __restrict__ frag2,
    const float* __restrict__ b1, const float* __restrict__ b2,
    const float* __restrict__ probs, int s,
    float* __restrict__ trace, float* __restrict__ finalOut, int writeFinal) {
    __shared__ float tile[TILE_F];      // [ry 0..3][c 0..9][col -1..64]

    int tid = threadIdx.x;
    int blk = blockIdx.x;               // 4096 blocks
    int b  = blk >> 5;                  // batch image
    int hb = (blk & 31) << 1;           // first of the two grid rows

    // ---- stage zero-padded halo: rows hb-1..hb+2, 10 ch, cols -1..64 ----
    const float* gsrc = gin + (size_t)b * gin_bstride;
    for (int i = tid; i < TILE_F; i += 256) {
        int cc = i % 66;                // padded column index
        int t  = i / 66;
        int c  = t % 10;
        int ry = t / 10;
        int row = hb + ry - 1;
        int col = cc - 1;
        float val = 0.f;
        if (row >= 0 && row < 64 && col >= 0 && col < 64)
            val = gsrc[(size_t)c * 4096 + (size_t)row * 64 + col];
        tile[i] = val;
    }
    __syncthreads();

    int lane = tid & 31;
    int widx = tid >> 5;
    int m16 = lane & 15, g = lane >> 4;
    int rr = widx >> 2;                 // row within the pair (0/1)
    int w0 = (widx & 3) << 4;           // col-tile base
    int h  = hb + rr;
    int base_col = w0 + m16;            // LDS padded-column base

    // im2col B-fragments from LDS; K = 10*dx + c, K = 2v + 16g + h2.
    // Channel/tap offsets collapse to a v_cndmask of two constants.
    BF16Frag Bp[3];
#pragma unroll
    for (int dy = 0; dy < 3; ++dy) {
        int rbase = (rr + dy) * 660 + base_col;   // (ry*10)*66 + col base
#pragma unroll
        for (int v = 0; v < 8; ++v) {
            float x[2];
#pragma unroll
            for (int h2 = 0; h2 < 2; ++h2) {
                const int Klo = 2 * v + h2;            // lane group 0 K
                const int Khi = Klo + 16;              // lane group 1 K
                const int cdLo = (Klo % 10) * 66 + (Klo / 10);
                const int cdHi = (Khi < 30) ? ((Khi % 10) * 66 + (Khi / 10))
                                            : cdLo;    // harmless dup
                int cd = g ? cdHi : cdLo;
                float val = tile[rbase + cd];
                if (Khi >= 30 && g) val = 0.f;         // K=30,31 zero pad
                x[h2] = val;
            }
            Bp[dy].u[v] = pk_bf16(x[0], x[1]);
        }
    }

    // program probabilities for this (batch, step): 4 x b128
    float pv[16];
    {
        const float4* pp = (const float4*)(probs + (b * 8 + s) * 16);
#pragma unroll
        for (int i = 0; i < 4; ++i) {
            float4 t = pp[i];
            pv[4 * i + 0] = t.x; pv[4 * i + 1] = t.y;
            pv[4 * i + 2] = t.z; pv[4 * i + 3] = t.w;
        }
    }

    v8f outAcc = (v8f)0.f;

    for (int e = 0; e < 16; ++e) {
        // conv1: two 16-outch tiles, 3 K-chunks each
        v8f acc0 = (v8f)0.f, acc1 = (v8f)0.f;
#pragma unroll
        for (int dy = 0; dy < 3; ++dy) {
            BF16Frag a0, a1;
            const uint4* p0 = (const uint4*)(frag1 + ((e * 3 + dy) * 2 + 0) * 256 + lane * 8);
            a0.q[0] = p0[0]; a0.q[1] = p0[1];
            const uint4* p1 = (const uint4*)(frag1 + ((e * 3 + dy) * 2 + 1) * 256 + lane * 8);
            a1.q[0] = p1[0]; a1.q[1] = p1[1];
            acc0 = wmma_bf16(a0.v, Bp[dy].v, acc0);
            acc1 = wmma_bf16(a1.v, Bp[dy].v, acc1);
        }

        // conv1 biases via 4 x global_load_b128 (b1 is float4-aligned here)
        float4 bA = ((const float4*)(b1 + e * 32 + 8 * g))[0];
        float4 bB = ((const float4*)(b1 + e * 32 + 8 * g))[1];
        float4 bC = ((const float4*)(b1 + e * 32 + 16 + 8 * g))[0];
        float4 bD = ((const float4*)(b1 + e * 32 + 16 + 8 * g))[1];
        float bias0[8] = { bA.x, bA.y, bA.z, bA.w, bB.x, bB.y, bB.z, bB.w };
        float bias1[8] = { bC.x, bC.y, bC.z, bC.w, bD.x, bD.y, bD.z, bD.w };

        // bias + ReLU, repack as conv2 A-fragment (pure per-lane)
        float hv[16];
#pragma unroll
        for (int j = 0; j < 8; ++j) {
            hv[j]     = fmaxf(acc0[j] + bias0[j], 0.f);
            hv[8 + j] = fmaxf(acc1[j] + bias1[j], 0.f);
        }
        BF16Frag a2;
#pragma unroll
        for (int v = 0; v < 8; ++v) a2.u[v] = pk_bf16(hv[2 * v], hv[2 * v + 1]);

        // conv2 (1x1): single WMMA, C initialized with output bias
        float bb2 = (m16 < 10) ? b2[e * 10 + m16] : 0.f;
        v8f acc2 = (v8f)bb2;
        BF16Frag bw;
        const uint4* p2 = (const uint4*)(frag2 + e * 256 + lane * 8);
        bw.q[0] = p2[0]; bw.q[1] = p2[1];
        acc2 = wmma_bf16(a2.v, bw.v, acc2);

        if (m16 < 10) {
            float pe = pv[e];
#pragma unroll
            for (int r = 0; r < 8; ++r) outAcc[r] += pe * fast_tanh(acc2[r]);
        }
    }

    if (m16 < 10) {
        // D2 lane layout: n = lane%16 (channel), pos = r + 8*(lane/16)
        size_t tidx = ((size_t)(b * 8 + s) * 10 + m16) * 4096 + (size_t)h * 64 + w0 + 8 * g;
        float4 o0 = { outAcc[0], outAcc[1], outAcc[2], outAcc[3] };
        float4 o1 = { outAcc[4], outAcc[5], outAcc[6], outAcc[7] };
        *(float4*)(trace + tidx)     = o0;
        *(float4*)(trace + tidx + 4) = o1;
        if (writeFinal) {
            size_t fidx = ((size_t)b * 10 + m16) * 4096 + (size_t)h * 64 + w0 + 8 * g;
            *(float4*)(finalOut + fidx)     = o0;
            *(float4*)(finalOut + fidx + 4) = o1;
        }
    }
}

// ---------------------------------------------------------------------------
extern "C" void kernel_launch(void* const* d_in, const int* in_sizes, int n_in,
                              void* d_out, int out_size, void* d_ws, size_t ws_size,
                              hipStream_t stream) {
    const float* features   = (const float*)d_in[0];
    const float* input_grid = (const float*)d_in[1];
    const float* Wg1 = (const float*)d_in[2];
    const float* bg1 = (const float*)d_in[3];
    const float* Wg2 = (const float*)d_in[4];
    const float* bg2 = (const float*)d_in[5];
    const float* W1  = (const float*)d_in[6];
    const float* b1  = (const float*)d_in[7];
    const float* W2  = (const float*)d_in[8];
    const float* b2  = (const float*)d_in[9];

    float* outF      = (float*)d_out;
    float* final_out = outF;                        // [128,10,64,64]
    float* probs_out = outF + 5242880;              // [128,8,16]
    float* trace_out = outF + 5242880 + 16384;      // [128,8,10,64,64]

    float* ws     = (float*)d_ws;
    float* g      = ws;                             // 65536
    float* hbuf   = ws + 65536;                     // 131072
    float* logits = ws + 196608;                    // 16384
    float* probs  = ws + 212992;                    // 16384
    unsigned int* frag1 = (unsigned int*)(ws + 229376); // 24576 dwords
    unsigned int* frag2 = (unsigned int*)(ws + 253952); // 4096 dwords

    pool_kernel<<<256, 256, 0, stream>>>(features, g);
    prep_frags<<<112, 256, 0, stream>>>(W1, W2, frag1, frag2);

    // Program generator: GEMM1 (+GELU), GEMM2, softmax
    gemm_bf16_kernel<<<64, 256, 0, stream>>>(g, Wg1, bg1, hbuf, 128, 1024, 512, 1, 512);
    gemm_bf16_kernel<<<8, 256, 0, stream>>>(hbuf, Wg2, bg2, logits, 128, 128, 1024, 0, 64);
    softmax_kernel<<<4, 256, 0, stream>>>(logits, probs, probs_out);

    // 8 sequential program steps; each reads previous grid, writes trace slice
    for (int s = 0; s < 8; ++s) {
        const float* gin = (s == 0) ? input_grid
                                    : (const float*)(trace_out + (size_t)(s - 1) * 40960);
        long bstride = (s == 0) ? 40960L : 327680L;
        step_kernel<<<4096, 256, 0, stream>>>(gin, bstride, frag1, frag2, b1, b2,
                                              probs, s, trace_out, final_out,
                                              (s == 7) ? 1 : 0);
    }
}
